// APPNPModel_16776142258480
// MI455X (gfx1250) — matrix operand order repaired and checked by reading, not compiled, not run on hardware
//
#include <hip/hip_runtime.h>

// ---------------------------------------------------------------------------
// Types for WMMA
// ---------------------------------------------------------------------------
typedef __attribute__((ext_vector_type(16))) __bf16 v16bf;
typedef __attribute__((ext_vector_type(8)))  float  v8f;

__device__ __forceinline__ unsigned short f32_to_bf16(float f) {
    unsigned int u = __float_as_uint(f);
    // round to nearest even
    unsigned int r = u + 0x7FFFu + ((u >> 16) & 1u);
    return (unsigned short)(r >> 16);
}

__device__ __forceinline__ unsigned int pack_bf16x2(float lo, float hi) {
    return (unsigned int)f32_to_bf16(lo) | ((unsigned int)f32_to_bf16(hi) << 16);
}

// ---------------------------------------------------------------------------
// Pack W [512 x 128] f32 (row-major, K x N) into bf16 B-fragment order:
//   frag f = kt*8 + nt (kt: K tiles of 32, nt: N tiles of 16)
//   out[(f*32 + lane)*16 + 2v + p] = W[k, n]
//   n = nt*16 + lane%16,  k = kt*32 + (lane/16)*8 + (e/8)*16 + e%8  (e = 2v + p)
// (CDNA5 ISA 7.12.2: 16-bit B 32x16 layout — lane = column, K pairs per VGPR)
// ---------------------------------------------------------------------------
__global__ void pack_w_kernel(const float* __restrict__ W,
                              unsigned short* __restrict__ out) {
    int t = blockIdx.x * blockDim.x + threadIdx.x;   // 0 .. 65535
    int e    = t & 15;
    int lane = (t >> 4) & 31;
    int nt   = (t >> 9) & 7;
    int kt   = t >> 12;
    int v = e >> 1;
    int p = e & 1;
    int n = nt * 16 + (lane & 15);
    int k = kt * 32 + (lane >> 4) * 8 + (e >> 3) * 16 + (e & 7);
    out[t] = f32_to_bf16(W[(size_t)k * 128 + n]);
}

// ---------------------------------------------------------------------------
// GEMM: C[M x 128] = A[M x 512] * W[512 x 128] + bias
// block = 128 threads (4 waves), tile M=64 (16 rows/wave), N=128 (8 WMMA tiles)
// Software-pipelined: next A chunk prefetched (global f32) while the current
// 8-WMMA chain runs; all 8 B fragments preloaded so WMMAs issue back-to-back.
// ---------------------------------------------------------------------------
__global__ void gemm_bf16_kernel(const float* __restrict__ A,
                                 const unsigned short* __restrict__ Bp,
                                 const float* __restrict__ bias,
                                 float* __restrict__ C, int M) {
    __shared__ __align__(32) unsigned short Alds[4][32][16];   // 4 KB

    const int tid  = threadIdx.x;
    const int wave = tid >> 5;
    const int lane = tid & 31;
    const int m0   = blockIdx.x * 64;

    v8f acc[8];
#pragma unroll
    for (int i = 0; i < 8; ++i)
        acc[i] = (v8f){0.f, 0.f, 0.f, 0.f, 0.f, 0.f, 0.f, 0.f};

    // staging: thread owns row r = tid/2, K-group hh = tid%2 (16 consecutive k)
    const int r  = tid >> 1;        // 0..63
    const int hh = tid & 1;         // 0,1
    int row = m0 + r;
    if (row >= M) row = M - 1;      // clamp: garbage rows only hit guarded stores
    const float4* src4 = (const float4*)(A + (size_t)row * 512 + hh * 16);

    unsigned int* dstLo = (unsigned int*)&Alds[r >> 4][(r & 15)][0];
    unsigned int* dstHi = (unsigned int*)&Alds[r >> 4][(r & 15) + 16][0];

    const unsigned short* bbase = Bp + (size_t)lane * 16;

    // prologue: fetch A chunk for kt = 0 (16 f32 = 4 x b128, coalesced)
    float4 af[4];
#pragma unroll
    for (int i = 0; i < 4; ++i) af[i] = src4[i];

    for (int kt = 0; kt < 16; ++kt) {
        // convert + store current A chunk into LDS in A-fragment order:
        // lane L (M=L%16), element 2v+p holds K=(v/4)*16+(L/16)*8+(v%4)*2+p
        const float* a = (const float*)af;
#pragma unroll
        for (int j = 0; j < 4; ++j) {
            int v = hh * 4 + j;
            dstLo[v] = pack_bf16x2(a[2 * j], a[2 * j + 1]);
            dstHi[v] = pack_bf16x2(a[8 + 2 * j], a[9 + 2 * j]);
        }
        __syncthreads();

        // prefetch next A chunk (overlaps with B loads + WMMA chain below)
        float4 anext[4];
        if (kt < 15) {
#pragma unroll
            for (int i = 0; i < 4; ++i) anext[i] = src4[(kt + 1) * 8 + i];
        }

        // A fragment: one aligned 32B LDS read per lane
        v16bf afrag = *(const v16bf*)&Alds[wave][lane][0];

        // preload all 8 B fragments -> one load clause, then 8 WMMAs b2b
        v16bf bfrag[8];
#pragma unroll
        for (int nt = 0; nt < 8; ++nt)
            bfrag[nt] = *(const v16bf*)&bbase[((size_t)kt * 8 + nt) * 512];

#pragma unroll
        for (int nt = 0; nt < 8; ++nt)
            acc[nt] = __builtin_amdgcn_wmma_f32_16x16x32_bf16(
                false, afrag, false, bfrag[nt], (short)0, acc[nt], false, false);

        __syncthreads();
#pragma unroll
        for (int i = 0; i < 4; ++i) af[i] = anext[i];
    }

    // C/D layout: lane L element v -> m = v + 8*(L/16), n = L%16
    const int mw   = m0 + wave * 16 + 8 * (lane >> 4);
    const int ncol = lane & 15;
#pragma unroll
    for (int nt = 0; nt < 8; ++nt) {
        int n = nt * 16 + ncol;
        float bn = bias[n];
#pragma unroll
        for (int v = 0; v < 8; ++v) {
            int m = mw + v;
            if (m < M) C[(size_t)m * 128 + n] = acc[nt][v] + bn;
        }
    }
}

// ---------------------------------------------------------------------------
// APPNP support kernels (fp32, bandwidth-bound)
// ---------------------------------------------------------------------------
__global__ void init_deg_kernel(float* __restrict__ deg, int N) {
    int i = blockIdx.x * blockDim.x + threadIdx.x;
    if (i < N) deg[i] = 1.0f;   // self-loop
}

__global__ void edge_deg_kernel(const int* __restrict__ ei,
                                float* __restrict__ deg, int E) {
    int e = blockIdx.x * blockDim.x + threadIdx.x;
    if (e < E) {
        int c = ei[E + e];      // target
        atomicAdd(&deg[c], 1.0f);
    }
}

__global__ void dinv_kernel(const float* __restrict__ deg,
                            float* __restrict__ dinv, int N) {
    int i = blockIdx.x * blockDim.x + threadIdx.x;
    if (i < N) dinv[i] = rsqrtf(deg[i]);   // deg >= 1 always
}

// agg[i,:] = dinv[i]^2 * h[i,:]   (self-loop message; also zero-initializes agg)
__global__ void seed_kernel(const float* __restrict__ dinv,
                            const float* __restrict__ h,
                            float* __restrict__ agg, int total) {
    int g = blockIdx.x * blockDim.x + threadIdx.x;
    if (g < total) {
        float di = dinv[g >> 7];
        agg[g] = di * di * h[g];
    }
}

// agg[col,:] += dinv[row]*dinv[col] * h[row,:]  (128 threads per edge)
__global__ void edge_prop_kernel(const int* __restrict__ ei,
                                 const float* __restrict__ dinv,
                                 const float* __restrict__ h,
                                 float* __restrict__ agg, int E) {
    int e = blockIdx.x * 2 + (threadIdx.x >> 7);
    int d = threadIdx.x & 127;
    if (e >= E) return;
    int r = ei[e];
    int c = ei[E + e];
    float nrm = dinv[r] * dinv[c];
    atomicAdd(&agg[(size_t)c * 128 + d], nrm * h[(size_t)r * 128 + d]);
}

// out = 0.9*agg + 0.1*h0
__global__ void combine_kernel(const float* __restrict__ agg,
                               const float* __restrict__ h0,
                               float* __restrict__ out, int total) {
    int g = blockIdx.x * blockDim.x + threadIdx.x;
    if (g < total) out[g] = 0.9f * agg[g] + 0.1f * h0[g];
}

// ---------------------------------------------------------------------------
// Launcher
// ---------------------------------------------------------------------------
extern "C" void kernel_launch(void* const* d_in, const int* in_sizes, int n_in,
                              void* d_out, int out_size, void* d_ws, size_t ws_size,
                              hipStream_t stream) {
    const float* x   = (const float*)d_in[0];   // [50000, 512]
    const int*   ei  = (const int*)d_in[1];     // [2, E]
    const float* q   = (const float*)d_in[2];   // [100000, 512]
    const float* Wt  = (const float*)d_in[3];   // [512, 128]
    const float* bt  = (const float*)d_in[4];   // [128]
    const float* Wq  = (const float*)d_in[5];   // [512, 128]
    const float* bq  = (const float*)d_in[6];   // [128]
    float* out = (float*)d_out;

    const int N  = in_sizes[0] / 512;           // 50000
    const int NQ = in_sizes[2] / 512;           // 100000
    const int E  = in_sizes[1] / 2;             // 800000
    const int ND = N * 128;

    // workspace layout
    char* ws = (char*)d_ws;
    unsigned short* pWt = (unsigned short*)ws;              // 65536 bf16
    unsigned short* pWq = pWt + 65536;                      // 65536 bf16
    float* deg  = (float*)(ws + 262144);                    // N
    float* dinv = deg + 50048;
    float* h0   = dinv + 50048;                             // N*128
    float* h1   = h0 + (size_t)ND;
    float* agg  = h1 + (size_t)ND;

    float* ques_out = out;                  // [NQ,128]
    float* h_out    = out + (size_t)NQ * 128;

    // 1) pack weights into bf16 WMMA B-fragment order
    pack_w_kernel<<<256, 256, 0, stream>>>(Wt, pWt);
    pack_w_kernel<<<256, 256, 0, stream>>>(Wq, pWq);

    // 2) encoders via bf16 WMMA GEMM
    gemm_bf16_kernel<<<(N + 63) / 64, 128, 0, stream>>>(x, pWt, bt, h0, N);
    gemm_bf16_kernel<<<(NQ + 63) / 64, 128, 0, stream>>>(q, pWq, bq, ques_out, NQ);

    // 3) GCN normalization
    init_deg_kernel<<<(N + 255) / 256, 256, 0, stream>>>(deg, N);
    edge_deg_kernel<<<(E + 255) / 256, 256, 0, stream>>>(ei, deg, E);
    dinv_kernel<<<(N + 255) / 256, 256, 0, stream>>>(deg, dinv, N);

    // 4) APPNP, K = 2
    // step 1: h0 -> h1
    seed_kernel<<<(ND + 255) / 256, 256, 0, stream>>>(dinv, h0, agg, ND);
    edge_prop_kernel<<<(E + 1) / 2, 256, 0, stream>>>(ei, dinv, h0, agg, E);
    combine_kernel<<<(ND + 255) / 256, 256, 0, stream>>>(agg, h0, h1, ND);
    // step 2: h1 -> d_out
    seed_kernel<<<(ND + 255) / 256, 256, 0, stream>>>(dinv, h1, agg, ND);
    edge_prop_kernel<<<(E + 1) / 2, 256, 0, stream>>>(ei, dinv, h1, agg, E);
    combine_kernel<<<(ND + 255) / 256, 256, 0, stream>>>(agg, h0, h_out, ND);
}
